// DeformableMHA_38225208934794
// MI455X (gfx1250) — matrix operand order repaired and checked by reading, not compiled, not run on hardware
//
#include <hip/hip_runtime.h>
#include <hip/hip_bf16.h>

// ---------------- problem constants (fixed by reference) ----------------
#define B_   8
#define K_   900
#define C_   256
#define NC_  6
#define H_   32
#define W_   88
#define NH_  8
#define NP_  4
#define HD_  32
#define HW_  (H_ * W_)          // 2816
#define MQC_ (B_ * K_ * NC_)    // 43200 rows of query_ (b,k,cam)
#define MQ_  (B_ * K_)          // 7200 rows of query
#define MF_  (NC_ * B_ * HW_)   // 135168 rows of featT (cam,b,pixel)

typedef float  v8f   __attribute__((ext_vector_type(8)));
typedef __bf16 v16bf __attribute__((ext_vector_type(16)));

__device__ __forceinline__ unsigned short f2bf(float f) {
  unsigned u = __builtin_bit_cast(unsigned, f);
  u += 0x7FFFu + ((u >> 16) & 1u);            // round-to-nearest-even
  return (unsigned short)(u >> 16);
}
__device__ __forceinline__ float bf2f(unsigned short h) {
  return __builtin_bit_cast(float, ((unsigned)h) << 16);
}

union Frag { v16bf v; uint4 u[2]; };

// ---------------- f32 -> bf16 convert ----------------
__global__ void cvt_bf16_kernel(const float* __restrict__ in,
                                unsigned short* __restrict__ out, int n) {
  int i = blockIdx.x * blockDim.x + threadIdx.x;
  if (i < n) out[i] = f2bf(in[i]);
}

// ---------------- features [cb][C][HW] f32 -> featT [cb][HW][C] bf16 ----------------
__global__ __launch_bounds__(256) void transpose_feat_kernel(
    const float* __restrict__ feat, unsigned short* __restrict__ featT) {
  __shared__ float tile[32][33];
  int tx = threadIdx.x & 31;
  int ty = threadIdx.x >> 5;           // 0..7
  int cb = blockIdx.z;                 // 0..47
  int p0 = blockIdx.x * 32;            // pixel tile
  int c0 = blockIdx.y * 32;            // channel tile
  const float* src = feat + ((size_t)cb * C_ + c0) * HW_ + p0;
#pragma unroll
  for (int i = 0; i < 4; ++i) {
    int c = ty + i * 8;
    tile[c][tx] = src[(size_t)c * HW_ + tx];     // coalesced over pixels
  }
  __syncthreads();
  unsigned short* dst = featT + ((size_t)cb * HW_ + p0) * C_ + c0;
#pragma unroll
  for (int i = 0; i < 4; ++i) {
    int pl = ty + i * 8;
    dst[(size_t)pl * C_ + tx] = f2bf(tile[tx][pl]);  // coalesced over channels
  }
}

// ---------------- camera projection: px[cam][b][k][2] ----------------
__global__ void px_kernel(const float* __restrict__ points,
                          const float* __restrict__ cam_M,
                          const float* __restrict__ cam_b,
                          float* __restrict__ px) {
  int i = blockIdx.x * blockDim.x + threadIdx.x;
  if (i >= NC_ * B_ * K_) return;
  int cam = i / (B_ * K_);
  int r   = i % (B_ * K_);
  const float* p = points + (size_t)r * 3;
  const float* M = cam_M + cam * 6;
  const float* cb = cam_b + cam * 2;
#pragma unroll
  for (int j = 0; j < 2; ++j) {
    float s = M[j * 3 + 0] * p[0] + M[j * 3 + 1] * p[1] + M[j * 3 + 2] * p[2] + cb[j];
    float sg = 1.0f / (1.0f + __expf(-s));
    px[(size_t)i * 2 + j] = sg * (j == 0 ? (float)H_ : (float)W_);
  }
}

// ---------------- camf sample + query add -> bf16 rows [(b,k,cam) x C] ----------------
__global__ __launch_bounds__(256) void camf_kernel(
    const float* __restrict__ query, const float* __restrict__ px,
    const unsigned short* __restrict__ featT, unsigned short* __restrict__ qc) {
  int wave = blockIdx.x * 8 + (threadIdx.x >> 5);
  if (wave >= MQC_) return;
  int lane = threadIdx.x & 31;
  int cam = wave % NC_;
  int bk  = wave / NC_;                 // b*K + k
  int b   = bk / K_;
  int k   = bk % K_;
  const float* pxp = px + ((size_t)((cam * B_ + b) * K_ + k)) * 2;
  float ix = pxp[0] * ((float)(W_ - 1) / (float)H_);   // gx -> width axis
  float iy = pxp[1] * ((float)(H_ - 1) / (float)W_);   // gy -> height axis
  float x0f = floorf(ix), y0f = floorf(iy);
  float fx = ix - x0f, fy = iy - y0f;
  int x0 = (int)x0f, y0 = (int)y0f;
  size_t pidx[4]; float wgt[4]; int ok[4];
#pragma unroll
  for (int cy = 0; cy < 2; ++cy)
#pragma unroll
    for (int cx = 0; cx < 2; ++cx) {
      int xx = x0 + cx, yy = y0 + cy;
      int ci = cy * 2 + cx;
      ok[ci]  = (xx >= 0 && xx < W_ && yy >= 0 && yy < H_);
      wgt[ci] = (cx ? fx : 1.0f - fx) * (cy ? fy : 1.0f - fy);
      pidx[ci] = (size_t)(yy * W_ + xx) * C_;
    }
  size_t fb = ((size_t)(cam * B_ + b)) * HW_ * C_;
  const float* qrow = query + (size_t)bk * C_;
  unsigned short* orow = qc + (size_t)wave * C_;
#pragma unroll
  for (int j = 0; j < 8; ++j) {
    int c = j * 32 + lane;
    float acc = qrow[c];
#pragma unroll
    for (int ci = 0; ci < 4; ++ci)
      if (ok[ci]) acc += wgt[ci] * bf2f(featT[fb + pidx[ci] + c]);
    orow[c] = f2bf(acc);
  }
}

// ---------------- register-blocked NT WMMA GEMM ----------------
// out[m,n] = sum_k A[m,k]*Bm[n,k] + bias[n]; A,Bm bf16 row-major (K contig).
// One wave computes a 32x64 block (2 M-tiles x 4 N-tiles): per K-step-32 it
// loads 2 A-frags + 4 B-frags (12 x B128) and issues 8 WMMAs (1.5 loads/WMMA).
template <bool OUT_BF16>
__global__ __launch_bounds__(128) void gemm_nt_wmma(
    const unsigned short* __restrict__ A, const unsigned short* __restrict__ Bm,
    const float* __restrict__ bias, void* __restrict__ Out,
    int M, int N, int Kd) {
  int wave = blockIdx.x * 4 + (threadIdx.x >> 5);
  int ntn = N >> 6;                     // groups of 4 N-tiles
  int total = (M >> 5) * ntn;
  if (wave >= total) return;
  int tm = wave / ntn, tn = wave % ntn;
  int lane = threadIdx.x & 31;
  int aSel = (lane & 16) ? 8 : 0;       // A: hi lanes hold K {8..15, 24..31}
  int bSel = (lane & 16) ? 8 : 0;       // B: hi lanes hold K {8..15, 24..31}
  const unsigned short* arow0 = A + (size_t)(tm * 32 + (lane & 15)) * Kd;
  const unsigned short* arow1 = arow0 + (size_t)16 * Kd;
  const unsigned short* brow  = Bm + (size_t)(tn * 64 + (lane & 15)) * Kd;
  v8f acc[2][4];
#pragma unroll
  for (int mi = 0; mi < 2; ++mi)
#pragma unroll
    for (int j = 0; j < 4; ++j) acc[mi][j] = (v8f){0.f,0.f,0.f,0.f,0.f,0.f,0.f,0.f};

  for (int kk = 0; kk < Kd; kk += 32) {
    Frag fa0, fa1, fb[4];
    const uint4* pa0 = (const uint4*)(arow0 + kk + aSel);
    fa0.u[0] = pa0[0];                  // K base+0..7
    fa0.u[1] = pa0[2];                  // K base+16..23
    const uint4* pa1 = (const uint4*)(arow1 + kk + aSel);
    fa1.u[0] = pa1[0];
    fa1.u[1] = pa1[2];
#pragma unroll
    for (int j = 0; j < 4; ++j) {
      const uint4* pb = (const uint4*)(brow + (size_t)j * 16 * Kd + kk + bSel);
      fb[j].u[0] = pb[0];               // K base+0..7
      fb[j].u[1] = pb[2];               // K base+16..23
    }
#pragma unroll
    for (int j = 0; j < 4; ++j) {
      acc[0][j] = __builtin_amdgcn_wmma_f32_16x16x32_bf16(
          false, fa0.v, false, fb[j].v, (short)0, acc[0][j], false, false);
      acc[1][j] = __builtin_amdgcn_wmma_f32_16x16x32_bf16(
          false, fa1.v, false, fb[j].v, (short)0, acc[1][j], false, false);
    }
  }
#pragma unroll
  for (int mi = 0; mi < 2; ++mi) {
    int rbase = tm * 32 + mi * 16 + ((lane & 16) ? 8 : 0);
#pragma unroll
    for (int j = 0; j < 4; ++j) {
      int ncol = tn * 64 + j * 16 + (lane & 15);
      float bv = bias[ncol];
#pragma unroll
      for (int g = 0; g < 8; ++g) {
        float v = acc[mi][j][g] + bv;
        size_t oi = (size_t)(rbase + g) * N + ncol;
        if (OUT_BF16) ((unsigned short*)Out)[oi] = f2bf(v);
        else          ((float*)Out)[oi] = v;
      }
    }
  }
}

// ---------------- fused deformable sampling + attention ----------------
// wave per (b,h,k), lane = d. Writes weighted in flat [b][h][k][d] order (the
// reference's "faithful" scrambled reshape), bf16 for the output projection.
__global__ __launch_bounds__(256) void attn_kernel(
    const float* __restrict__ qbuf, const float* __restrict__ offs,
    const float* __restrict__ px, const unsigned short* __restrict__ kfT,
    const unsigned short* __restrict__ vfT, unsigned short* __restrict__ wout) {
  int wave = blockIdx.x * 8 + (threadIdx.x >> 5);
  if (wave >= B_ * NH_ * K_) return;
  int lane = threadIdx.x & 31;
  int b = wave / (NH_ * K_);
  int h = (wave / K_) % NH_;
  int k = wave % K_;
  float qd = qbuf[((size_t)(b * K_ + k)) * C_ + h * HD_ + lane];
  float logit[NC_ * NP_];
  float val[NC_ * NP_];
#pragma unroll
  for (int cam = 0; cam < NC_; ++cam) {
    const float* pxp = px + ((size_t)((cam * B_ + b) * K_ + k)) * 2;
    float px0 = pxp[0], px1 = pxp[1];
    const float* op = offs + ((size_t)((b * K_ + k) * NC_ + cam)) * (2 * NH_ * NP_)
                    + h * (NP_ * 2);
    size_t fbase = ((size_t)(cam * B_ + b)) * HW_ * C_ + h * HD_ + lane;
#pragma unroll
    for (int pt = 0; pt < NP_; ++pt) {
      int li = cam * NP_ + pt;
      float pc0 = px0 + op[pt * 2 + 0] * (1.0f / (float)H_);
      float pc1 = px1 + op[pt * 2 + 1] * (1.0f / (float)W_);
      float ix = pc0 * ((float)(W_ - 1) / (float)H_);
      float iy = pc1 * ((float)(H_ - 1) / (float)W_);
      float x0f = floorf(ix), y0f = floorf(iy);
      float fx = ix - x0f, fy = iy - y0f;
      int x0 = (int)x0f, y0 = (int)y0f;
      float kacc = 0.f, vacc = 0.f;
#pragma unroll
      for (int cy = 0; cy < 2; ++cy)
#pragma unroll
        for (int cx = 0; cx < 2; ++cx) {
          int xx = x0 + cx, yy = y0 + cy;
          float wgt = (cx ? fx : 1.0f - fx) * (cy ? fy : 1.0f - fy);
          if (xx >= 0 && xx < W_ && yy >= 0 && yy < H_) {
            size_t idx = fbase + (size_t)(yy * W_ + xx) * C_;
            kacc += wgt * bf2f(kfT[idx]);
            vacc += wgt * bf2f(vfT[idx]);
          }
        }
      float t = qd * kacc;                      // wave32 dot over hd
#pragma unroll
      for (int m = 16; m >= 1; m >>= 1) t += __shfl_xor(t, m, 32);
      logit[li] = t * 0.17677669529663687f;     // 1/sqrt(32)
      val[li] = vacc;
    }
  }
  float mx = logit[0];
#pragma unroll
  for (int l = 1; l < NC_ * NP_; ++l) mx = fmaxf(mx, logit[l]);
  float s = 0.f;
#pragma unroll
  for (int l = 0; l < NC_ * NP_; ++l) { logit[l] = __expf(logit[l] - mx); s += logit[l]; }
  float inv = 1.0f / s;
  float acc = 0.f;
#pragma unroll
  for (int l = 0; l < NC_ * NP_; ++l) acc += logit[l] * val[l];
  wout[((size_t)((b * NH_ + h) * K_ + k)) * HD_ + lane] = f2bf(acc * inv);
}

// ---------------- launch ----------------
extern "C" void kernel_launch(void* const* d_in, const int* in_sizes, int n_in,
                              void* d_out, int out_size, void* d_ws, size_t ws_size,
                              hipStream_t stream) {
  const float* query    = (const float*)d_in[0];
  const float* points   = (const float*)d_in[1];
  const float* features = (const float*)d_in[2];
  const float* cam_M    = (const float*)d_in[3];
  const float* cam_b    = (const float*)d_in[4];
  const float* W_off    = (const float*)d_in[5];
  const float* b_off    = (const float*)d_in[6];
  const float* W_q      = (const float*)d_in[7];
  const float* b_q      = (const float*)d_in[8];
  const float* W_k      = (const float*)d_in[9];
  const float* b_k      = (const float*)d_in[10];
  const float* W_v      = (const float*)d_in[11];
  const float* b_v      = (const float*)d_in[12];
  const float* W_o      = (const float*)d_in[13];
  const float* b_o      = (const float*)d_in[14];
  float* out = (float*)d_out;

  char* w = (char*)d_ws;
  size_t o = 0;
  auto take = [&](size_t bytes) { size_t cur = o; o += (bytes + 255) & ~(size_t)255; return cur; };
  unsigned short* ws_qc    = (unsigned short*)(w + take((size_t)MQC_ * C_ * 2)); // query_ bf16
  unsigned short* ws_qbf   = (unsigned short*)(w + take((size_t)MQ_ * C_ * 2));  // query bf16
  unsigned short* ws_wq    = (unsigned short*)(w + take((size_t)C_ * C_ * 2));
  unsigned short* ws_wk    = (unsigned short*)(w + take((size_t)C_ * C_ * 2));
  unsigned short* ws_wv    = (unsigned short*)(w + take((size_t)C_ * C_ * 2));
  unsigned short* ws_wo    = (unsigned short*)(w + take((size_t)C_ * C_ * 2));
  unsigned short* ws_woff  = (unsigned short*)(w + take((size_t)64 * C_ * 2));
  unsigned short* ws_featT = (unsigned short*)(w + take((size_t)MF_ * C_ * 2));  // [cam,b,p,c]
  unsigned short* ws_kfT   = (unsigned short*)(w + take((size_t)MF_ * C_ * 2));
  unsigned short* ws_vfT   = (unsigned short*)(w + take((size_t)MF_ * C_ * 2));
  float* ws_px   = (float*)(w + take((size_t)NC_ * B_ * K_ * 2 * 4));
  float* ws_off  = (float*)(w + take((size_t)MQC_ * 64 * 4));
  float* ws_qprj = (float*)(w + take((size_t)MQ_ * C_ * 4));
  unsigned short* ws_wsum = (unsigned short*)(w + take((size_t)MQ_ * C_ * 2));
  (void)ws_size; (void)in_sizes; (void)n_in; (void)out_size;

  // 1) bf16 conversions (weights + query)
  cvt_bf16_kernel<<<(MQ_ * C_ + 255) / 256, 256, 0, stream>>>(query, ws_qbf, MQ_ * C_);
  cvt_bf16_kernel<<<(C_ * C_ + 255) / 256, 256, 0, stream>>>(W_q, ws_wq, C_ * C_);
  cvt_bf16_kernel<<<(C_ * C_ + 255) / 256, 256, 0, stream>>>(W_k, ws_wk, C_ * C_);
  cvt_bf16_kernel<<<(C_ * C_ + 255) / 256, 256, 0, stream>>>(W_v, ws_wv, C_ * C_);
  cvt_bf16_kernel<<<(C_ * C_ + 255) / 256, 256, 0, stream>>>(W_o, ws_wo, C_ * C_);
  cvt_bf16_kernel<<<(64 * C_ + 255) / 256, 256, 0, stream>>>(W_off, ws_woff, 64 * C_);

  // 2) transpose features -> [cam,b,pixel,channel] bf16 (K-contiguous for GEMM,
  //    channel-coalesced for gathers)
  transpose_feat_kernel<<<dim3(HW_ / 32, C_ / 32, NC_ * B_), 256, 0, stream>>>(
      features, ws_featT);

  // 3) camera projection
  px_kernel<<<(NC_ * B_ * K_ + 255) / 256, 256, 0, stream>>>(points, cam_M, cam_b, ws_px);

  // 4) camf sample + query add -> query_ rows (bf16)
  camf_kernel<<<MQC_ / 8, 256, 0, stream>>>(query, ws_px, ws_featT, ws_qc);

  // 5) offsets GEMM: [43200,256] x [64,256]^T  (waves = (M/32)*(N/64) = 1350)
  {
    int waves = (MQC_ / 32) * (64 / 64);
    gemm_nt_wmma<false><<<(waves + 3) / 4, 128, 0, stream>>>(
        ws_qc, ws_woff, b_off, ws_off, MQC_, 64, C_);
  }

  // 6) K / V 1x1-conv GEMMs: [135168,256] x [256,256]^T -> bf16 (dominant FLOPs)
  {
    int waves = (MF_ / 32) * (C_ / 64);   // 16896
    gemm_nt_wmma<true><<<(waves + 3) / 4, 128, 0, stream>>>(
        ws_featT, ws_wk, b_k, ws_kfT, MF_, C_, C_);
    gemm_nt_wmma<true><<<(waves + 3) / 4, 128, 0, stream>>>(
        ws_featT, ws_wv, b_v, ws_vfT, MF_, C_, C_);
  }

  // 7) Q projection: [7200,256] x [256,256]^T -> f32
  {
    int waves = (MQ_ / 32) * (C_ / 64);   // 900
    gemm_nt_wmma<false><<<(waves + 3) / 4, 128, 0, stream>>>(
        ws_qbf, ws_wq, b_q, ws_qprj, MQ_, C_, C_);
  }

  // 8) fused deformable sampling + attention -> weighted (bf16, scrambled layout)
  attn_kernel<<<(B_ * NH_ * K_) / 8, 256, 0, stream>>>(
      ws_qprj, ws_off, ws_px, ws_kfT, ws_vfT, ws_wsum);

  // 9) output projection: [7200,256] x [256,256]^T -> d_out f32
  {
    int waves = (MQ_ / 32) * (C_ / 64);
    gemm_nt_wmma<false><<<(waves + 3) / 4, 128, 0, stream>>>(
        ws_wsum, ws_wo, b_o, out, MQ_, C_, C_);
  }
}
